// UNMASMixerSum_4982162063530
// MI455X (gfx1250) — matrix operand order repaired and checked by reading, not compiled, not run on hardware
//
#include <hip/hip_runtime.h>
#include <hip/hip_bf16.h>

// ---------------------------------------------------------------------------
// Types for CDNA5 WMMA (gfx1250, wave32)
// ---------------------------------------------------------------------------
typedef __bf16 bf16_t;
typedef bf16_t         v16bf __attribute__((ext_vector_type(16)));
typedef float          v8f   __attribute__((ext_vector_type(8)));
typedef unsigned short u16;
typedef u16            v16u  __attribute__((ext_vector_type(16)));
typedef unsigned int   v4u   __attribute__((ext_vector_type(4)));
typedef unsigned int   v2u   __attribute__((ext_vector_type(2)));
typedef float          v4f   __attribute__((ext_vector_type(4)));

union Frag  { v16u u; v16bf b; v4u q[2]; };
union Pack4 { u16 h[4]; v2u v; };

// Native bf16 conversion: lowers to hardware v_cvt on gfx1250 (RNE).
__device__ __forceinline__ u16 f2bf(float f) {
  union { bf16_t b; u16 u; } c;
  c.b = (bf16_t)f;
  return c.u;
}

__device__ __forceinline__ v8f v8zero() {
  v8f z;
#pragma unroll
  for (int i = 0; i < 8; ++i) z[i] = 0.0f;
  return z;
}

__device__ __forceinline__ v8f wmma_bf16(const Frag& a, const Frag& b, v8f c) {
  // D = A(16x32 bf16) * B(32x16 bf16) + C(16x16 f32)
  return __builtin_amdgcn_wmma_f32_16x16x32_bf16(false, a.b, false, b.b,
                                                 (short)0, c, false, false);
}

// A-fragment (16x32 bf16) from row-major f32 source, converted in registers.
// Lane layout (ISA 7.12.2): lanes 0-15 M=0..15 hold K {0..7,16..23},
// lanes 16-31 same M hold K {8..15,24..31}.
__device__ __forceinline__ Frag loadA_f32(const float* X, long long row0, int ld, int k0) {
  const int lane = threadIdx.x & 31;
  const int m = lane & 15, hi = lane >> 4;
  const float* p = X + (long long)(row0 + m) * ld + k0 + 8 * hi;
  const v4f* p4 = (const v4f*)p;
  v4f xs[4] = { p4[0], p4[1], p4[4], p4[5] };   // K +0..7 and +16..23 (per half-wave)
  Frag f;
#pragma unroll
  for (int t = 0; t < 4; ++t)
#pragma unroll
    for (int j = 0; j < 4; ++j) f.u[t * 4 + j] = f2bf(xs[t][j]);
  return f;
}

// A-fragment from LDS bf16 row-major [16 x ld]
__device__ __forceinline__ Frag loadA_lds(const u16* L, int ld, int k0) {
  const int lane = threadIdx.x & 31;
  const int m = lane & 15, hi = lane >> 4;
  const u16* p = L + m * ld + k0 + 8 * hi;
  Frag f;
  f.q[0] = ((const v4u*)p)[0];          // halves 0..7
  f.q[1] = *((const v4u*)(p + 16));     // halves 8..15
  return f;
}

// B-fragment (32x16 bf16) from row-major bf16 weights [K x ld]:
// lane holds column n0+lane%16, K {0..7,16..23} (lanes 0-15) / {8..15,24..31} (lanes 16-31).
__device__ __forceinline__ Frag loadB(const u16* W, int ld, int k0, int n0) {
  const int lane = threadIdx.x & 31;
  const int n = lane & 15, hi = lane >> 4;
  const u16* p = W + (long long)(k0 + 8 * hi) * ld + n0 + n;
  Frag f;
#pragma unroll
  for (int s = 0; s < 8; ++s) {
    f.u[s]     = p[(long long)s * ld];
    f.u[8 + s] = p[(long long)(16 + s) * ld];
  }
  return f;
}

// ---------------------------------------------------------------------------
// Fused MLP tiles (one wave computes a 16-row tile). Activation modes:
//   MODE 0 = none, 1 = abs, 2 = relu
// ---------------------------------------------------------------------------

// X(global f32) @ W0 + b0 -> relu -> bf16 LDS [16 x H]   (obs layer 1)
template <int K, int H>
__device__ __forceinline__ void mm_g_to_lh(const float* X, long long row0, int ldx,
                                           const u16* W0, const float* b0, u16* Lh) {
  constexpr int KS = K / 32, NF = H / 16;
  v8f acc[NF];
#pragma unroll
  for (int i = 0; i < NF; ++i) acc[i] = v8zero();
#pragma unroll 1
  for (int ks = 0; ks < KS; ++ks) {
    Frag A = loadA_f32(X, row0, ldx, ks * 32);
#pragma unroll
    for (int nf = 0; nf < NF; ++nf) {
      Frag B = loadB(W0, H, ks * 32, nf * 16);
      acc[nf] = wmma_bf16(A, B, acc[nf]);
    }
  }
  const int lane = threadIdx.x & 31;
  const int c0 = lane & 15, hi = lane >> 4;
#pragma unroll
  for (int nf = 0; nf < NF; ++nf) {
    const int col = nf * 16 + c0;
    const float bv = b0[col];
#pragma unroll
    for (int r = 0; r < 8; ++r) {
      float h = acc[nf][r] + bv;
      h = h > 0.0f ? h : 0.0f;
      Lh[(r + 8 * hi) * H + col] = f2bf(h);
    }
  }
}

// A(bf16 LDS [16 x ldA]) @ W0 + b0 -> relu -> bf16 LDS [16 x H]   (state layer 1)
template <int K, int H>
__device__ __forceinline__ void mm_l_to_lh(const u16* LA, int ldA,
                                           const u16* W0, const float* b0, u16* Lh) {
  constexpr int KS = K / 32, NF = H / 16;
  v8f acc[NF];
#pragma unroll
  for (int i = 0; i < NF; ++i) acc[i] = v8zero();
#pragma unroll 1
  for (int ks = 0; ks < KS; ++ks) {
    Frag A = loadA_lds(LA, ldA, ks * 32);
#pragma unroll
    for (int nf = 0; nf < NF; ++nf) {
      Frag B = loadB(W0, H, ks * 32, nf * 16);
      acc[nf] = wmma_bf16(A, B, acc[nf]);
    }
  }
  const int lane = threadIdx.x & 31;
  const int c0 = lane & 15, hi = lane >> 4;
#pragma unroll
  for (int nf = 0; nf < NF; ++nf) {
    const int col = nf * 16 + c0;
    const float bv = b0[col];
#pragma unroll
    for (int r = 0; r < 8; ++r) {
      float h = acc[nf][r] + bv;
      h = h > 0.0f ? h : 0.0f;
      Lh[(r + 8 * hi) * H + col] = f2bf(h);
    }
  }
}

// A(bf16 LDS [16 x ldA]) @ W + b -> act -> f32 out (global or LDS)
template <int K, int N2, int MODE>
__device__ __forceinline__ void mm_l_to_f32(const u16* LA, int ldA,
                                            const u16* W, const float* b,
                                            float* out, long long orow0, int ldo) {
  constexpr int KS = K / 32, NF = N2 / 16;
  v8f acc[NF];
#pragma unroll
  for (int i = 0; i < NF; ++i) acc[i] = v8zero();
#pragma unroll 1
  for (int ks = 0; ks < KS; ++ks) {
    Frag A = loadA_lds(LA, ldA, ks * 32);
#pragma unroll
    for (int nf = 0; nf < NF; ++nf) {
      Frag B = loadB(W, N2, ks * 32, nf * 16);
      acc[nf] = wmma_bf16(A, B, acc[nf]);
    }
  }
  const int lane = threadIdx.x & 31;
  const int c0 = lane & 15, hi = lane >> 4;
#pragma unroll
  for (int nf = 0; nf < NF; ++nf) {
    const int col = nf * 16 + c0;
    const float bv = b[col];
#pragma unroll
    for (int r = 0; r < 8; ++r) {
      float v = acc[nf][r] + bv;
      if (MODE == 1) v = fabsf(v);
      if (MODE == 2) v = v > 0.0f ? v : 0.0f;
      out[(orow0 + r + 8 * hi) * (long long)ldo + col] = v;
    }
  }
}

// ---------------------------------------------------------------------------
// Kernels
// ---------------------------------------------------------------------------
__global__ void cvt_kernel(const float* src, u16* dst, int n) {
  int i = blockIdx.x * blockDim.x + threadIdx.x;
  if (i < n) dst[i] = f2bf(src[i]);
}

#define WAVES_ST 4
__global__ void __launch_bounds__(128)
state_kernel(const float* st,
             const u16* w_sh0,  const float* b_sh0,  const u16* w_sh1,  const float* b_sh1,
             const u16* w_hwk0, const float* b_hwk0, const u16* w_hwk1, const float* b_hwk1,
             const u16* w_hw10, const float* b_hw10, const u16* w_hw11, const float* b_hw11,
             const u16* w_hwq0, const float* b_hwq0, const u16* w_hwq1, const float* b_hwq1,
             const u16* w_hb1,  const float* b_hb1,
             const u16* w_hbq0, const float* b_hbq0, const float* hbq_w1, const float* hbq_b1,
             const u16* w_hbk0, const float* b_hbk0, const float* hbk_w1, const float* hbk_b1,
             const u16* w_v0,   const float* b_v0,   const float* v_w1,   const float* v_b1,
             float* w1A, float* wqA, float* wkA, float* b1v, float* scal) {
  extern __shared__ char smem[];
  const int wid  = threadIdx.x >> 5;
  const int lane = threadIdx.x & 31;
  u16*   Lst  = (u16*)smem + wid * (16 * 512);                                          // 16KB/wave
  u16*   Lh   = (u16*)(smem + WAVES_ST * 16 * 512 * 2) + wid * (16 * 256);              // 8KB/wave
  float* scrA = (float*)(smem + WAVES_ST * (16 * 512 * 2 + 16 * 256 * 2)) + wid * (16 * 64);              // 4KB
  float* scrB = (float*)(smem + WAVES_ST * (16 * 512 * 2 + 16 * 256 * 2 + 16 * 64 * 4)) + wid * (16 * 128); // 8KB
  const long long tile = (long long)blockIdx.x * WAVES_ST + wid;
  const long long r0 = tile * 16;

  // Stage the 16x512 state tile once as bf16 into LDS (read st from HBM exactly once).
  for (int i = lane; i < (16 * 512) / 4; i += 32) {
    const int e = i * 4;
    const int m = e >> 9, c = e & 511;
    v4f x = *(const v4f*)(st + (r0 + m) * 512 + c);
    Pack4 p;
#pragma unroll
    for (int j = 0; j < 4; ++j) p.h[j] = f2bf(x[j]);
    *(v2u*)(Lst + m * 512 + c) = p.v;
  }

  // state_hidden: 512 -> relu 256 -> 64 (linear) -> LDS
  mm_l_to_lh<512, 256>(Lst, 512, w_sh0, b_sh0, Lh);
  mm_l_to_f32<256, 64, 0>(Lh, 256, w_sh1, b_sh1, scrA, 0, 64);

  // hyper_w_k: 512 -> relu 256 -> 128, |.| -> LDS
  mm_l_to_lh<512, 256>(Lst, 512, w_hwk0, b_hwk0, Lh);
  mm_l_to_f32<256, 128, 1>(Lh, 256, w_hwk1, b_hwk1, scrB, 0, 128);

  // |w_k|[:, :64] -> global; s_k = sum(|w_k|[:, 64:] * sh)
  for (int i = lane; i < 16 * 64; i += 32) {
    int m = i >> 6, c = i & 63;
    wkA[(r0 + m) * 64 + c] = scrB[m * 128 + c];
  }
  float s_k = 0.0f;
  if (lane < 16) {
#pragma unroll 4
    for (int c = 0; c < 64; ++c) s_k += scrB[lane * 128 + 64 + c] * scrA[lane * 64 + c];
  }

  // hyper_w_1 -> |.| -> w1A ; hyper_w_q -> |.| -> wqA
  mm_l_to_lh<512, 256>(Lst, 512, w_hw10, b_hw10, Lh);
  mm_l_to_f32<256, 64, 1>(Lh, 256, w_hw11, b_hw11, w1A, r0, 64);
  mm_l_to_lh<512, 256>(Lst, 512, w_hwq0, b_hwq0, Lh);
  mm_l_to_f32<256, 64, 1>(Lh, 256, w_hwq1, b_hwq1, wqA, r0, 64);

  // hyper_b_1 (linear 512->64)
  mm_l_to_f32<512, 64, 0>(Lst, 512, w_hb1, b_hb1, b1v, r0, 64);

  // hyper_b_q / hyper_b_k / V : relu(512->64) then 64-dot
  mm_l_to_f32<512, 64, 2>(Lst, 512, w_hbq0, b_hbq0, scrA, 0, 64);
  float bq = 0.0f;
  if (lane < 16) {
    for (int c = 0; c < 64; ++c) bq += scrA[lane * 64 + c] * hbq_w1[c];
    bq += hbq_b1[0];
  }
  mm_l_to_f32<512, 64, 2>(Lst, 512, w_hbk0, b_hbk0, scrA, 0, 64);
  float bk = 0.0f;
  if (lane < 16) {
    for (int c = 0; c < 64; ++c) bk += scrA[lane * 64 + c] * hbk_w1[c];
    bk += hbk_b1[0];
  }
  mm_l_to_f32<512, 64, 2>(Lst, 512, w_v0, b_v0, scrA, 0, 64);
  if (lane < 16) {
    float vv = 0.0f;
    for (int c = 0; c < 64; ++c) vv += scrA[lane * 64 + c] * v_w1[c];
    vv += v_b1[0];
    const long long n = r0 + lane;
    scal[n * 4 + 0] = bq;
    scal[n * 4 + 1] = bk + s_k;
    scal[n * 4 + 2] = vv;
    scal[n * 4 + 3] = 0.0f;
  }
}

#define WAVES_OBS 8
__global__ void __launch_bounds__(256)
obs_kernel(const float* obs, const u16* W0, const float* b0,
           const u16* W1, const float* b1, float* oh) {
  extern __shared__ char smem[];
  const int wid = threadIdx.x >> 5;
  u16* Lh = (u16*)smem + wid * (16 * 256);
  const long long tile = (long long)blockIdx.x * WAVES_OBS + wid;
  const long long r0 = tile * 16;
  mm_g_to_lh<256, 256>(obs, r0, 256, W0, b0, Lh);      // relu(obs @ oh_w0 + b0)
  mm_l_to_f32<256, 64, 0>(Lh, 256, W1, b1, oh, r0, 64); // @ oh_w1 + b1
}

__device__ __forceinline__ float elu1(float x) { return x > 0.0f ? x : (__expf(x) - 1.0f); }

__global__ void __launch_bounds__(128)
combine_kernel(const float* qs, const float* alive,
               const float* w1A, const float* b1v, const float* wqA, const float* wkA,
               const float* scal, const float* oh, float* out, int nrows) {
  const int wid  = threadIdx.x >> 5;
  const int lane = threadIdx.x & 31;
  const long long n = (long long)blockIdx.x * 4 + wid;
  if (n >= nrows) return;
  float q = 0.0f, k = 0.0f;
  if (lane < 16) {
    const long long row = n * 16 + lane;           // agent = lane
    const v4f* oh4 = (const v4f*)(oh + row * 64);
    const v4f* w14 = (const v4f*)(w1A + n * 64);
    const v4f* b14 = (const v4f*)(b1v + n * 64);
    const v4f* wq4 = (const v4f*)(wqA + n * 64);
    const v4f* wk4 = (const v4f*)(wkA + n * 64);
    const float qa = qs[row];
    const float al = alive[row];
    float kd = 0.0f;
#pragma unroll 4
    for (int eb = 0; eb < 16; ++eb) {
      v4f w1v = w14[eb], b1e = b14[eb], wqv = wq4[eb], wkv = wk4[eb], ov = oh4[eb];
#pragma unroll
      for (int j = 0; j < 4; ++j) {
        q  += elu1(qa * w1v[j] + b1e[j]) * wqv[j];
        kd += ov[j] * wkv[j];
      }
    }
    q += scal[n * 4 + 0];
    k  = __expf(kd + scal[n * 4 + 1]);
    q *= al;
    k *= al;
  }
  float qk = q * k;
#pragma unroll
  for (int off = 16; off > 0; off >>= 1) {
    k  += __shfl_xor(k, off);
    qk += __shfl_xor(qk, off);
  }
  if (lane == 0) out[n] = qk / (k + 0.01f) + scal[n * 4 + 2];
}

// ---------------------------------------------------------------------------
// Host launch
// ---------------------------------------------------------------------------
extern "C" void kernel_launch(void* const* d_in, const int* in_sizes, int n_in,
                              void* d_out, int out_size, void* d_ws, size_t ws_size,
                              hipStream_t stream) {
  (void)in_sizes; (void)n_in; (void)out_size; (void)ws_size;
  const float* qs    = (const float*)d_in[0];
  const float* st    = (const float*)d_in[1];
  const float* obs   = (const float*)d_in[2];
  const float* alive = (const float*)d_in[3];

  const int N = 64 * 512;        // BS*T = 32768
  const int NOBS = N * 16;       // 524288 obs rows

  char* ws = (char*)d_ws;
  size_t off = 0;
  auto alloc = [&](size_t bytes) -> void* {
    void* p = ws + off;
    off = (off + bytes + 255) & ~(size_t)255;
    return p;
  };
  auto cvt = [&](int idx, int n) -> u16* {
    u16* dst = (u16*)alloc((size_t)n * 2);
    cvt_kernel<<<(n + 255) / 256, 256, 0, stream>>>((const float*)d_in[idx], dst, n);
    return dst;
  };

  // bf16 weight copies
  u16* hw1w0 = cvt(4, 512 * 256);
  u16* hw1w1 = cvt(6, 256 * 64);
  u16* hwqw0 = cvt(8, 512 * 256);
  u16* hwqw1 = cvt(10, 256 * 64);
  u16* hwkw0 = cvt(12, 512 * 256);
  u16* hwkw1 = cvt(14, 256 * 128);
  u16* ohw0  = cvt(16, 256 * 256);
  u16* ohw1  = cvt(18, 256 * 64);
  u16* shw0  = cvt(20, 512 * 256);
  u16* shw1  = cvt(22, 256 * 64);
  u16* hb1w  = cvt(24, 512 * 64);
  u16* hbqw0 = cvt(26, 512 * 64);
  u16* hbkw0 = cvt(30, 512 * 64);
  u16* vw0   = cvt(34, 512 * 64);

  float* w1A  = (float*)alloc((size_t)N * 64 * 4);
  float* wqA  = (float*)alloc((size_t)N * 64 * 4);
  float* wkA  = (float*)alloc((size_t)N * 64 * 4);
  float* b1v  = (float*)alloc((size_t)N * 64 * 4);
  float* scal = (float*)alloc((size_t)N * 4 * 4);
  float* oh   = (float*)alloc((size_t)NOBS * 64 * 4);

  const size_t sh_state =
      WAVES_ST * (16 * 512 * 2 + 16 * 256 * 2 + 16 * 64 * 4 + 16 * 128 * 4); // 144KB
  state_kernel<<<(N / 16) / WAVES_ST, WAVES_ST * 32, sh_state, stream>>>(
      st,
      shw0,  (const float*)d_in[21], shw1,  (const float*)d_in[23],
      hwkw0, (const float*)d_in[13], hwkw1, (const float*)d_in[15],
      hw1w0, (const float*)d_in[5],  hw1w1, (const float*)d_in[7],
      hwqw0, (const float*)d_in[9],  hwqw1, (const float*)d_in[11],
      hb1w,  (const float*)d_in[25],
      hbqw0, (const float*)d_in[27], (const float*)d_in[28], (const float*)d_in[29],
      hbkw0, (const float*)d_in[31], (const float*)d_in[32], (const float*)d_in[33],
      vw0,   (const float*)d_in[35], (const float*)d_in[36], (const float*)d_in[37],
      w1A, wqA, wkA, b1v, scal);

  const size_t sh_obs = WAVES_OBS * 16 * 256 * 2; // 64KB
  obs_kernel<<<(NOBS / 16) / WAVES_OBS, WAVES_OBS * 32, sh_obs, stream>>>(
      obs, ohw0, (const float*)d_in[17], ohw1, (const float*)d_in[19], oh);

  combine_kernel<<<(N + 3) / 4, 128, 0, stream>>>(
      qs, alive, w1A, b1v, wqA, wkA, scal, oh, (float*)d_out, N);
}
